// HeteNet_12171937317349
// MI455X (gfx1250) — matrix operand order repaired and checked by reading, not compiled, not run on hardware
//
#include <hip/hip_runtime.h>
#include <hip/hip_bf16.h>
#include <math.h>

// ---------------- problem constants ----------------
#define T_DIM 64
#define A_DIM 48
#define TA    3072          // T*A agents
#define E_DIM 22
#define RAW   512
#define H     256
#define H2    512           // 2*H
#define CM    1536          // 2*H + 2*RAW
#define NACT  21
#define NTP   3

typedef __bf16 bf16_t;
typedef __bf16 v16bf __attribute__((ext_vector_type(16)));
typedef float  v8f   __attribute__((ext_vector_type(8)));

union FragBF { uint4 u4[2]; v16bf v; };

// per-type bf16 transposed-weight offsets (elements) inside d_ws weight block
#define OFF_W1T   0          // [H ][RAW]
#define OFF_W2T   131072     // [H ][H  ]
#define OFF_WCfT  196608     // [H ][H2 ]
#define OFF_WMfT  327680     // [H ][CM ]
#define OFF_WChT  720896     // [H ][H2 ]
#define OFF_WMhT  851968     // [H ][CM ]
#define OFF_L1T   1245184    // [H ][H2 ]
#define OFF_L2T   1376256    // [128][H ]
#define OFF_L3T   1409024    // [32 ][128] (21 rows valid, rest zero)
#define OFF_V1T   1413120    // [H ][H2 ]
#define OFF_V2T   1544192    // [16 ][H ] (1 row valid)
#define PER_TYPE  1548288

static __device__ __forceinline__ v8f zero8() {
    v8f z = {0.f,0.f,0.f,0.f,0.f,0.f,0.f,0.f};
    return z;
}

// ---- WMMA fragment helpers (layouts per CDNA5 ISA 7.12.2) ----
// A 16x32 bf16: lanes 0-15 row M=lane, K {0..7,16..23}; lanes 16-31 K {8..15,24..31}
static __device__ __forceinline__ FragBF load_fragA(const bf16_t* A, int ld, int row0,
                                                    int k0, int lane) {
    int half = lane >> 4, m = lane & 15;
    const bf16_t* p = A + (size_t)(row0 + m) * ld + k0 + half * 8;
    FragBF f;
    f.u4[0] = *(const uint4*)p;          // K = k0+half*8 .. +7
    f.u4[1] = *(const uint4*)(p + 16);   // K = k0+16+half*8 .. +7
    return f;
}
// A rows gathered per-agent from a global feature table X[agent][K]
static __device__ __forceinline__ FragBF load_fragA_rows(const bf16_t* __restrict__ X, int K,
                                                         const int* rows, int k0, int lane) {
    int half = lane >> 4, m = lane & 15;
    const bf16_t* p = X + (size_t)rows[m] * K + k0 + half * 8;
    FragBF f;
    f.u4[0] = *(const uint4*)p;
    f.u4[1] = *(const uint4*)(p + 16);
    return f;
}
// B 32x16 bf16 from transposed weights BT[N][K]: lane = column n, lanes 0-15 K {0..7,16..23}; lanes 16-31 K {8..15,24..31}
static __device__ __forceinline__ FragBF load_fragB(const bf16_t* __restrict__ BT, int ldk,
                                                    int n0, int k0, int lane) {
    int half = lane >> 4, m = lane & 15;
    const bf16_t* p = BT + (size_t)(n0 + m) * ldk + k0 + half * 8;
    FragBF f;
    f.u4[0] = *(const uint4*)p;          // K = k0+half*8 .. +7
    f.u4[1] = *(const uint4*)(p + 16);   // K = k0+16+half*8 .. +7
    return f;
}

static __device__ __forceinline__ v8f wmma_bf16(FragBF a, FragBF b, v8f c) {
    return __builtin_amdgcn_wmma_f32_16x16x32_bf16(
        /*neg_a=*/false, a.v, /*neg_b=*/false, b.v,
        /*c_mod=*/(short)0, c, /*reuse_a=*/false, /*reuse_b=*/false);
}

// C/D 16x16 f32 layout: VGPR r -> row = half*8 + r, col = lane&15
static __device__ __forceinline__ void store_tile_bf(bf16_t* out, int ld, int row0, int n0,
                                                     v8f acc, const float* bias, int nbias,
                                                     bool relu, int lane) {
    int half = lane >> 4, m = lane & 15, n = n0 + m;
    float bv = (n < nbias) ? bias[n] : 0.f;
#pragma unroll
    for (int r = 0; r < 8; ++r) {
        float vv = acc[r] + bv;
        if (relu) vv = fmaxf(vv, 0.f);
        out[(size_t)(row0 + half * 8 + r) * ld + n] = (bf16_t)vv;
    }
}
static __device__ __forceinline__ void store_tile_f32(float* out, int ld, int row0, int n0,
                                                      v8f acc, const float* bias, int nbias,
                                                      bool relu, int lane) {
    int half = lane >> 4, m = lane & 15, n = n0 + m;
    float bv = (n < nbias) ? bias[n] : 0.f;
#pragma unroll
    for (int r = 0; r < 8; ++r) {
        float vv = acc[r] + bv;
        if (relu) vv = fmaxf(vv, 0.f);
        out[(size_t)(row0 + half * 8 + r) * ld + n] = vv;
    }
}

// ------------- K0: fp32 -> bf16 transposed weight conversion -------------
// src: [NTP][K][N] fp32, dst slot: [NTP stride PER_TYPE][NP][K] bf16 (rows >= N zeroed)
__global__ void k_cvt(const float* __restrict__ src, bf16_t* __restrict__ dst,
                      int K, int N, int NP) {
    int t = blockIdx.y;
    const float* s = src + (size_t)t * K * N;
    bf16_t* d = dst + (size_t)t * PER_TYPE;
    size_t tot = (size_t)NP * K;
    for (size_t idx = (size_t)blockIdx.x * blockDim.x + threadIdx.x; idx < tot;
         idx += (size_t)gridDim.x * blockDim.x) {
        int n = (int)(idx / K), k = (int)(idx % K);
        d[idx] = (n < N) ? (bf16_t)s[(size_t)k * N + n] : (bf16_t)0.f;
    }
}

// ------------- K1: per-agent entity MLP + pooling -> feature vectors -------------
__global__ __launch_bounds__(256) void k_embed(
    const float* __restrict__ obs, const int* __restrict__ hete,
    const unsigned char* __restrict__ dead,
    const float* __restrict__ b1, const float* __restrict__ b2,
    const bf16_t* __restrict__ wbf,
    bf16_t* __restrict__ xCf, bf16_t* __restrict__ xMf,
    bf16_t* __restrict__ xCh, bf16_t* __restrict__ xMh) {
    __shared__ __align__(16) bf16_t As[32 * RAW];  // 32KB, later aliased as Vs f32 32x256
    __shared__ __align__(16) bf16_t Hs[32 * H];    // 16KB
    __shared__ float s_den[2];
    __shared__ unsigned char s_dead[E_DIM];

    const int ta = blockIdx.x;
    const int tid = threadIdx.x;
    const int lane = tid & 31, w = tid >> 5;
    int t = hete[ta];
    if (t < 0) t = 0;
    if (t >= NTP) t = NTP - 1;
    const bf16_t* W1T = wbf + (size_t)t * PER_TYPE + OFF_W1T;
    const bf16_t* W2T = wbf + (size_t)t * PER_TYPE + OFF_W2T;
    const float* b1t = b1 + (size_t)t * H;
    const float* b2t = b2 + (size_t)t * H;

    if (tid < E_DIM) s_dead[tid] = dead[(size_t)ta * E_DIM + tid];
    __syncthreads();
    if (tid == 0) {
        float df = 0.f, dh = 0.f;
        for (int e = 1; e < 12; ++e)  df += s_dead[e] ? 0.f : 1.f;
        for (int e = 12; e < 22; ++e) dh += s_dead[e] ? 0.f : 1.f;
        s_den[0] = fmaxf(df, 1.f);
        s_den[1] = fmaxf(dh, 1.f);
    }
    // stage dead-masked obs as bf16, rows 22..31 zero-padded
    const float* obsrow = obs + (size_t)ta * E_DIM * RAW;
    for (int idx = tid; idx < 32 * RAW; idx += 256) {
        int e = idx >> 9, c = idx & (RAW - 1);
        float vv = 0.f;
        if (e < E_DIM && !s_dead[e]) vv = obsrow[(size_t)e * RAW + c];
        As[idx] = (bf16_t)vv;
    }
    __syncthreads();

    // GEMM1: Hs = relu(As(32x512) @ W1 + b1), bf16 WMMA, f32 accumulate
    {
        const int m0 = (w & 1) * 16;
        const int n0b = (w >> 1) * 64;
        v8f a0 = zero8(), a1 = zero8(), a2 = zero8(), a3 = zero8();
        for (int k0 = 0; k0 < RAW; k0 += 32) {
            FragBF a = load_fragA(As, RAW, m0, k0, lane);
            FragBF b;
            b = load_fragB(W1T, RAW, n0b + 0, k0, lane);  a0 = wmma_bf16(a, b, a0);
            b = load_fragB(W1T, RAW, n0b + 16, k0, lane); a1 = wmma_bf16(a, b, a1);
            b = load_fragB(W1T, RAW, n0b + 32, k0, lane); a2 = wmma_bf16(a, b, a2);
            b = load_fragB(W1T, RAW, n0b + 48, k0, lane); a3 = wmma_bf16(a, b, a3);
        }
        store_tile_bf(Hs, H, m0, n0b + 0, a0, b1t, H, true, lane);
        store_tile_bf(Hs, H, m0, n0b + 16, a1, b1t, H, true, lane);
        store_tile_bf(Hs, H, m0, n0b + 32, a2, b1t, H, true, lane);
        store_tile_bf(Hs, H, m0, n0b + 48, a3, b1t, H, true, lane);
    }
    __syncthreads();

    // z-features (zs + masked-mean pools of raw obs) while As is still live
    for (int c = tid; c < RAW; c += 256) {
        float zs = (float)As[c];  // entity row 0
        float sf = 0.f, sh = 0.f;
        for (int e = 1; e < 12; ++e)  sf += (float)As[e * RAW + c];
        for (int e = 12; e < 22; ++e) sh += (float)As[e * RAW + c];
        sf /= s_den[0];
        sh /= s_den[1];
        bf16_t zb = (bf16_t)zs;
        xMf[(size_t)ta * CM + 512 + c] = zb;
        xMf[(size_t)ta * CM + 1024 + c] = (bf16_t)sf;
        xMh[(size_t)ta * CM + 512 + c] = zb;
        xMh[(size_t)ta * CM + 1024 + c] = (bf16_t)sh;
    }
    __syncthreads();

    // GEMM2: Vs(f32, aliases As) = Hs(32x256) @ W2 + b2 (no relu)
    float* Vs = (float*)As;
    {
        const int m0 = (w & 1) * 16;
        const int n0b = (w >> 1) * 64;
        v8f a0 = zero8(), a1 = zero8(), a2 = zero8(), a3 = zero8();
        for (int k0 = 0; k0 < H; k0 += 32) {
            FragBF a = load_fragA(Hs, H, m0, k0, lane);
            FragBF b;
            b = load_fragB(W2T, H, n0b + 0, k0, lane);  a0 = wmma_bf16(a, b, a0);
            b = load_fragB(W2T, H, n0b + 16, k0, lane); a1 = wmma_bf16(a, b, a1);
            b = load_fragB(W2T, H, n0b + 32, k0, lane); a2 = wmma_bf16(a, b, a2);
            b = load_fragB(W2T, H, n0b + 48, k0, lane); a3 = wmma_bf16(a, b, a3);
        }
        store_tile_f32(Vs, H, m0, n0b + 0, a0, b2t, H, false, lane);
        store_tile_f32(Vs, H, m0, n0b + 16, a1, b2t, H, false, lane);
        store_tile_f32(Vs, H, m0, n0b + 32, a2, b2t, H, false, lane);
        store_tile_f32(Vs, H, m0, n0b + 48, a3, b2t, H, false, lane);
    }
    __syncthreads();

    // v-features: vs + masked-mean pools of v
    for (int c = tid; c < H; c += 256) {
        float vs = Vs[c];
        float pf = 0.f, ph = 0.f;
        for (int e = 1; e < 12; ++e)  pf += Vs[e * H + c];
        for (int e = 12; e < 22; ++e) ph += Vs[e * H + c];
        pf /= s_den[0];
        ph /= s_den[1];
        bf16_t vsb = (bf16_t)vs, pfb = (bf16_t)pf, phb = (bf16_t)ph;
        xCf[(size_t)ta * H2 + c] = vsb;       xCf[(size_t)ta * H2 + H + c] = pfb;
        xCh[(size_t)ta * H2 + c] = vsb;       xCh[(size_t)ta * H2 + H + c] = phb;
        xMf[(size_t)ta * CM + c] = vsb;       xMf[(size_t)ta * CM + H + c] = pfb;
        xMh[(size_t)ta * CM + c] = vsb;       xMh[(size_t)ta * CM + H + c] = phb;
    }
}

// ------------- K2: bucket agents by type (so head WMMA tiles share weights) -------------
__global__ void k_compact(const int* __restrict__ hete, int* __restrict__ counts,
                          int* __restrict__ lists) {
    __shared__ int c[NTP];
    int tid = threadIdx.x;
    if (tid < NTP) c[tid] = 0;
    __syncthreads();
    for (int i = tid; i < TA; i += blockDim.x) {
        int t = hete[i];
        if (t < 0) t = 0;
        if (t >= NTP) t = NTP - 1;
        int pos = atomicAdd(&c[t], 1);
        lists[(size_t)t * TA + pos] = i;
    }
    __syncthreads();
    if (tid < NTP) counts[tid] = c[tid];
}

// ------------- head GEMM helpers (16-row tiles) -------------
static __device__ __forceinline__ void gemm_gather(
    const bf16_t* __restrict__ X, int K, const int* rows,
    const bf16_t* __restrict__ BT, const float* __restrict__ bias, int nbias,
    bf16_t* out, int ldout, int Nout, bool relu, int tid) {
    int w = tid >> 5, lane = tid & 31;
    for (int n0 = w * 16; n0 < Nout; n0 += 128) {
        v8f acc = zero8();
        for (int k0 = 0; k0 < K; k0 += 32) {
            FragBF a = load_fragA_rows(X, K, rows, k0, lane);
            FragBF b = load_fragB(BT, K, n0, k0, lane);
            acc = wmma_bf16(a, b, acc);
        }
        store_tile_bf(out, ldout, 0, n0, acc, bias, nbias, relu, lane);
    }
}
static __device__ __forceinline__ void gemm_lds(
    const bf16_t* __restrict__ A, int K,
    const bf16_t* __restrict__ BT, const float* __restrict__ bias, int nbias,
    bf16_t* out, int ldout, int Nout, bool relu, int tid) {
    int w = tid >> 5, lane = tid & 31;
    for (int n0 = w * 16; n0 < Nout; n0 += 128) {
        v8f acc = zero8();
        for (int k0 = 0; k0 < K; k0 += 32) {
            FragBF a = load_fragA(A, K, 0, k0, lane);
            FragBF b = load_fragB(BT, K, n0, k0, lane);
            acc = wmma_bf16(a, b, acc);
        }
        store_tile_bf(out, ldout, 0, n0, acc, bias, nbias, relu, lane);
    }
}
static __device__ __forceinline__ void gemm_lds_f32(
    const bf16_t* __restrict__ A, int K,
    const bf16_t* __restrict__ BT, const float* __restrict__ bias, int nbias,
    float* out, int ldout, int Nout, int tid) {
    int w = tid >> 5, lane = tid & 31;
    for (int n0 = w * 16; n0 < Nout; n0 += 128) {
        v8f acc = zero8();
        for (int k0 = 0; k0 < K; k0 += 32) {
            FragBF a = load_fragA(A, K, 0, k0, lane);
            FragBF b = load_fragB(BT, K, n0, k0, lane);
            acc = wmma_bf16(a, b, acc);
        }
        store_tile_f32(out, ldout, 0, n0, acc, bias, nbias, false, lane);
    }
}

// ------------- K3: per-type 16-agent head tiles -------------
__global__ __launch_bounds__(256) void k_heads(
    const int* __restrict__ counts, const int* __restrict__ lists,
    const bf16_t* __restrict__ wbf,
    const bf16_t* __restrict__ xCf, const bf16_t* __restrict__ xMf,
    const bf16_t* __restrict__ xCh, const bf16_t* __restrict__ xMh,
    const float* __restrict__ bCf, const float* __restrict__ bMf,
    const float* __restrict__ bCh, const float* __restrict__ bMh,
    const float* __restrict__ bL1, const float* __restrict__ bL2,
    const float* __restrict__ bL3, const float* __restrict__ bV1,
    const float* __restrict__ bV2, void* __restrict__ outbuf) {
    const int type = blockIdx.y;
    const int n_t = counts[type];
    const int base = blockIdx.x * 16;
    if (base >= n_t) return;

    __shared__ int rows[16];
    __shared__ __align__(16) bf16_t vC[16 * H2];   // 16KB  [vfC | vhC]
    __shared__ __align__(16) bf16_t vM[16 * H2];   // 16KB  [vfM | vhM]
    __shared__ __align__(16) bf16_t hb[16 * H];    // 8KB   hL1 then hV1
    __shared__ __align__(16) bf16_t h2b[16 * 128]; // 4KB   hL2
    __shared__ __align__(16) float lg[16 * 32];    // logits (21 valid)
    __shared__ __align__(16) float vl[16 * 16];    // value tile (col 0 valid)

    const int tid = threadIdx.x;
    if (tid < 16) {
        int idx = base + tid;
        rows[tid] = lists[(size_t)type * TA + (idx < n_t ? idx : base)];
    }
    __syncthreads();

    const bf16_t* wt = wbf + (size_t)type * PER_TYPE;
    const float* bCft = bCf + (size_t)type * H;
    const float* bMft = bMf + (size_t)type * H;
    const float* bCht = bCh + (size_t)type * H;
    const float* bMht = bMh + (size_t)type * H;
    const float* bL1t = bL1 + (size_t)type * H;
    const float* bL2t = bL2 + (size_t)type * 128;
    const float* bL3t = bL3 + (size_t)type * NACT;
    const float* bV1t = bV1 + (size_t)type * H;
    const float* bV2t = bV2 + (size_t)type * 1;

    gemm_gather(xCf, H2, rows, wt + OFF_WCfT, bCft, H, vC, H2, H, true, tid);
    gemm_gather(xCh, H2, rows, wt + OFF_WChT, bCht, H, vC + H, H2, H, true, tid);
    gemm_gather(xMf, CM, rows, wt + OFF_WMfT, bMft, H, vM, H2, H, true, tid);
    gemm_gather(xMh, CM, rows, wt + OFF_WMhT, bMht, H, vM + H, H2, H, true, tid);
    __syncthreads();
    gemm_lds(vC, H2, wt + OFF_L1T, bL1t, H, hb, H, H, true, tid);
    __syncthreads();
    gemm_lds(hb, H, wt + OFF_L2T, bL2t, 128, h2b, 128, 128, true, tid);
    __syncthreads();
    gemm_lds_f32(h2b, 128, wt + OFF_L3T, bL3t, NACT, lg, 32, 32, tid);
    __syncthreads();
    gemm_lds(vM, H2, wt + OFF_V1T, bV1t, H, hb, H, H, true, tid);
    __syncthreads();
    gemm_lds_f32(hb, H, wt + OFF_V2T, bV2t, 1, vl, 16, 16, tid);
    __syncthreads();

    // per-row argmax / log-softmax / value epilogue
    if (tid < 16) {
        int idx = base + tid;
        if (idx < n_t) {
            int ag = rows[tid];
            float mx = -1e30f;
            int am = 0;
            for (int i = 0; i < NACT; ++i) {
                float v = lg[tid * 32 + i];
                if (v > mx) { mx = v; am = i; }
            }
            float s = 0.f;
            for (int i = 0; i < NACT; ++i) s += __expf(lg[tid * 32 + i] - mx);
            float alp = lg[tid * 32 + am] - mx - __logf(s);
            int* actp = (int*)outbuf;
            float* valp = (float*)outbuf + TA;
            float* alpp = (float*)outbuf + 2 * TA;
            actp[ag] = am;
            valp[ag] = vl[tid * 16];
            alpp[ag] = alp;
        }
    }
}

// ---------------- host launch ----------------
extern "C" void kernel_launch(void* const* d_in, const int* in_sizes, int n_in,
                              void* d_out, int out_size, void* d_ws, size_t ws_size,
                              hipStream_t stream) {
    (void)in_sizes; (void)n_in; (void)out_size;
    const float* obs = (const float*)d_in[0];
    const int* hete = (const int*)d_in[1];
    const unsigned char* dead = (const unsigned char*)d_in[2];  // jax bool -> 1 byte
    const float* W1 = (const float*)d_in[3],  *b1  = (const float*)d_in[4];
    const float* W2 = (const float*)d_in[5],  *b2  = (const float*)d_in[6];
    const float* WCf = (const float*)d_in[7], *bCf = (const float*)d_in[8];
    const float* WMf = (const float*)d_in[9], *bMf = (const float*)d_in[10];
    const float* WCh = (const float*)d_in[11],*bCh = (const float*)d_in[12];
    const float* WMh = (const float*)d_in[13],*bMh = (const float*)d_in[14];
    const float* L1 = (const float*)d_in[15], *bL1 = (const float*)d_in[16];
    const float* L2 = (const float*)d_in[17], *bL2 = (const float*)d_in[18];
    const float* L3 = (const float*)d_in[19], *bL3 = (const float*)d_in[20];
    const float* V1 = (const float*)d_in[21], *bV1 = (const float*)d_in[22];
    const float* V2 = (const float*)d_in[23], *bV2 = (const float*)d_in[24];

    // workspace layout: [bf16 weights NTP*PER_TYPE][xCf][xMf][xCh][xMh][counts/lists]
    const size_t WS_NEEDED =
        ((size_t)NTP * PER_TYPE + (size_t)TA * (H2 + CM + H2 + CM)) * 2 +
        (8 + (size_t)NTP * TA) * 4;
    if (ws_size < WS_NEEDED) return;  // insufficient scratch

    bf16_t* wbf = (bf16_t*)d_ws;
    bf16_t* xCf = wbf + (size_t)NTP * PER_TYPE;
    bf16_t* xMf = xCf + (size_t)TA * H2;
    bf16_t* xCh = xMf + (size_t)TA * CM;
    bf16_t* xMh = xCh + (size_t)TA * H2;
    int* counts = (int*)(xMh + (size_t)TA * CM);
    int* lists = counts + 8;

    dim3 blk(256);
    auto cvt = [&](const float* s, size_t off, int K, int N, int NP) {
        int blocks = (int)(((size_t)NP * K + 255) / 256);
        k_cvt<<<dim3(blocks, NTP), blk, 0, stream>>>(s, wbf + off, K, N, NP);
    };
    cvt(W1, OFF_W1T, RAW, H, H);
    cvt(W2, OFF_W2T, H, H, H);
    cvt(WCf, OFF_WCfT, H2, H, H);
    cvt(WMf, OFF_WMfT, CM, H, H);
    cvt(WCh, OFF_WChT, H2, H, H);
    cvt(WMh, OFF_WMhT, CM, H, H);
    cvt(L1, OFF_L1T, H2, H, H);
    cvt(L2, OFF_L2T, H, 128, 128);
    cvt(L3, OFF_L3T, 128, NACT, 32);
    cvt(V1, OFF_V1T, H2, H, H);
    cvt(V2, OFF_V2T, H, 1, 16);

    k_embed<<<dim3(TA), blk, 0, stream>>>(obs, hete, dead, b1, b2, wbf,
                                          xCf, xMf, xCh, xMh);
    k_compact<<<dim3(1), blk, 0, stream>>>(hete, counts, lists);
    k_heads<<<dim3((TA + 15) / 16, NTP), blk, 0, stream>>>(
        counts, lists, wbf, xCf, xMf, xCh, xMh,
        bCf, bMf, bCh, bMh, bL1, bL2, bL3, bV1, bV2, d_out);
}
